// MAMIL2D_26130581029415
// MI455X (gfx1250) — compile-verified, matrix operand order checked by reading, not run on hardware
//
#include <hip/hip_runtime.h>
#include <hip/hip_bf16.h>
#include <math.h>

// ---------------------------------------------------------------------------
// MAMIL2D on gfx1250 (MI455X).
//  - Heavy GEMMs: v_wmma_f32_16x16x32_f16, f32 accumulate.
//  - B tiles double-buffered in LDS via global_load_async_to_lds_b128
//    (ASYNCcnt) and consumed with ds_load_b128 -> B traffic shared by all
//    waves of the workgroup.  All 4 B sub-tiles are loaded into distinct
//    register groups before the 8-WMMA burst so only one dscnt wait per
//    K-step is exposed.
//  - O(N^2) attention computed sparsely over the 8-neighborhood.
// ---------------------------------------------------------------------------

typedef __attribute__((ext_vector_type(16))) _Float16 v16h;
typedef __attribute__((ext_vector_type(8)))  _Float16 v8h;
typedef __attribute__((ext_vector_type(8)))  float    v8f;

#define NINST 4096
#define LDIM  512
#define L2DIM 1024
#define EMBD  1728
#define NTPL  10
#define DDIM  128

// ---------------------------------------------------------------------------
// conv1 (36,3,4,4) + relu + maxpool2 : x(4096,3,32,32) -> h1(4096,36,14,14)
// ---------------------------------------------------------------------------
__global__ __launch_bounds__(256)
void k_conv1_pool(const float* __restrict__ x, const float* __restrict__ w,
                  const float* __restrict__ b, float* __restrict__ h1) {
    int idx = blockIdx.x * blockDim.x + threadIdx.x;
    const int total = NINST * 36 * 14 * 14;
    if (idx >= total) return;
    int px = idx % 14;
    int py = (idx / 14) % 14;
    int oc = (idx / 196) % 36;
    int n  = idx / (196 * 36);
    float bias = b[oc];
    float m = -3.4e38f;
#pragma unroll
    for (int dy = 0; dy < 2; ++dy) {
#pragma unroll
        for (int dx = 0; dx < 2; ++dx) {
            int y0 = 2 * py + dy;
            int x0 = 2 * px + dx;
            float s = bias;
#pragma unroll
            for (int ic = 0; ic < 3; ++ic) {
                const float* xp = x + (((size_t)n * 3 + ic) * 32 + y0) * 32 + x0;
                const float* wp = w + ((oc * 3 + ic) * 16);
#pragma unroll
                for (int ky = 0; ky < 4; ++ky) {
#pragma unroll
                    for (int kx = 0; kx < 4; ++kx)
                        s = fmaf(xp[ky * 32 + kx], wp[ky * 4 + kx], s);
                }
            }
            s = fmaxf(s, 0.0f);
            m = fmaxf(m, s);
        }
    }
    h1[idx] = m;
}

// ---------------------------------------------------------------------------
// conv2 (48,36,3,3) + relu + maxpool2 : h1(4096,36,14,14) -> h2f16(4096,1728)
// ---------------------------------------------------------------------------
__global__ __launch_bounds__(256)
void k_conv2_pool(const float* __restrict__ h1, const float* __restrict__ w,
                  const float* __restrict__ b, _Float16* __restrict__ h2h) {
    int idx = blockIdx.x * blockDim.x + threadIdx.x;
    const int total = NINST * 48 * 6 * 6;
    if (idx >= total) return;
    int px = idx % 6;
    int py = (idx / 6) % 6;
    int oc = (idx / 36) % 48;
    int n  = idx / (36 * 48);
    float bias = b[oc];
    float m = -3.4e38f;
#pragma unroll
    for (int dy = 0; dy < 2; ++dy) {
#pragma unroll
        for (int dx = 0; dx < 2; ++dx) {
            int y0 = 2 * py + dy;
            int x0 = 2 * px + dx;
            float s = bias;
            for (int ic = 0; ic < 36; ++ic) {
                const float* hp = h1 + (((size_t)n * 36 + ic) * 14 + y0) * 14 + x0;
                const float* wp = w + ((oc * 36 + ic) * 9);
#pragma unroll
                for (int ky = 0; ky < 3; ++ky) {
#pragma unroll
                    for (int kx = 0; kx < 3; ++kx)
                        s = fmaf(hp[ky * 14 + kx], wp[ky * 3 + kx], s);
                }
            }
            s = fmaxf(s, 0.0f);
            m = fmaxf(m, s);
        }
    }
    h2h[(size_t)n * EMBD + oc * 36 + py * 6 + px] = (_Float16)m;
}

// ---------------------------------------------------------------------------
// f32 -> f16 convert
// ---------------------------------------------------------------------------
__global__ __launch_bounds__(256)
void k_cvt_h(const float* __restrict__ src, _Float16* __restrict__ dst, int n) {
    int i = blockIdx.x * blockDim.x + threadIdx.x;
    if (i < n) dst[i] = (_Float16)src[i];
}

// ---------------------------------------------------------------------------
// WMMA NT-GEMM:  C(MxN) = act( A(MxK) * W(NxK)^T + bias )
// Block = 4 waves; each wave owns a 32x64 output tile (8 accumulators).
// B tile (64x32 halfs) double-buffered in LDS via async copy; all waves share.
// M%128==0, N%64==0, K%32==0.
// ---------------------------------------------------------------------------
__device__ __forceinline__ v16h ld16h(const _Float16* lo, const _Float16* hi) {
    union { v16h v; v8h h[2]; } u;
    u.h[0] = *(const v8h*)lo;
    u.h[1] = *(const v8h*)hi;
    return u.v;
}

// 16B memory -> LDS async copy (ASYNCcnt).  Low 32 bits of a generic pointer
// to a __shared__ object are the raw LDS byte address (aperture rule).
__device__ __forceinline__ void async_b128_to_lds(const _Float16* gsrc, _Float16* ldst) {
    unsigned int lds_addr = (unsigned int)(unsigned long long)(uintptr_t)ldst;
    unsigned long long gaddr = (unsigned long long)(uintptr_t)gsrc;
    asm volatile("global_load_async_to_lds_b128 %0, %1, off"
                 :: "v"(lds_addr), "v"(gaddr)
                 : "memory");
}
__device__ __forceinline__ void wait_asynccnt0() {
    asm volatile("s_wait_asynccnt 0" ::: "memory");
}

__global__ __launch_bounds__(128)
void k_gemm_nt(const _Float16* __restrict__ A, const _Float16* __restrict__ W,
               const float* __restrict__ bias, float* __restrict__ C,
               int M, int N, int K, int act) {
    __shared__ _Float16 Bs[2][64 * 32];   // 2 x 4KB double-buffered B tile

    const int lane = threadIdx.x & 31;
    const int wave = threadIdx.x >> 5;
    const int m0 = blockIdx.x * 128 + wave * 32;
    const int n0 = blockIdx.y * 64;
    const int row   = lane & 15;
    const int khalf = (lane >> 4) << 3;   // 0 or 8 (CDNA5 16-bit A/B layout)

    const _Float16* Ap0 = A + (size_t)(m0 + row) * K;
    const _Float16* Ap1 = A + (size_t)(m0 + 16 + row) * K;

    // cooperative async copy of B[k..k+31] strip for rows n0..n0+63:
    // 256 chunks of 16B, 128 threads -> 2 chunks/thread.
    auto copyB = [&](int kk, int buf) {
#pragma unroll
        for (int c0 = 0; c0 < 2; ++c0) {
            int c = threadIdx.x + c0 * 128;
            int n = c >> 2;
            int part = c & 3;
            const _Float16* g = W + (size_t)(n0 + n) * K + kk + part * 8;
            async_b128_to_lds(g, &Bs[buf][n * 32 + part * 8]);
        }
    };

    v8f acc[8];
#pragma unroll
    for (int s = 0; s < 8; ++s) acc[s] = (v8f){};

    copyB(0, 0);
    wait_asynccnt0();
    __syncthreads();

    int cur = 0;
    for (int k = 0; k < K; k += 32) {
        const bool more = (k + 32) < K;
        if (more) copyB(k + 32, cur ^ 1);

        const int k0 = k + khalf;
        v16h a0 = ld16h(Ap0 + k0, Ap0 + k0 + 16);
        v16h a1 = ld16h(Ap1 + k0, Ap1 + k0 + 16);
        if (more) {
            __builtin_prefetch(Ap0 + k0 + 32, 0, 1);   // global_prefetch_b8
            __builtin_prefetch(Ap1 + k0 + 32, 0, 1);
        }

        // Load all 4 B sub-tiles into distinct register groups first
        // (8x ds_load_b128, one dscnt wait), then 8 back-to-back WMMAs.
        v16h b[4];
#pragma unroll
        for (int s = 0; s < 4; ++s) {
            const _Float16* bp = &Bs[cur][(row + s * 16) * 32 + khalf];
            b[s] = ld16h(bp, bp + 16);
        }
#pragma unroll
        for (int s = 0; s < 4; ++s) {
            acc[s]     = __builtin_amdgcn_wmma_f32_16x16x32_f16(false, a0, false, b[s], (short)0, acc[s],     false, false);
            acc[s + 4] = __builtin_amdgcn_wmma_f32_16x16x32_f16(false, a1, false, b[s], (short)0, acc[s + 4], false, false);
        }
        if (more) wait_asynccnt0();
        __syncthreads();
        cur ^= 1;
    }

    // C/D layout: VGPR r -> M = r (+8 for lanes 16..31), N = lane&15.
    const int rowbase = m0 + ((lane >> 4) << 3);
    const int colbase = n0 + (lane & 15);
#pragma unroll
    for (int s = 0; s < 4; ++s) {
        const int cc = colbase + s * 16;
        const float bv = bias[cc];
#pragma unroll
        for (int r = 0; r < 8; ++r) {
            float v0 = acc[s][r] + bv;
            float v1 = acc[s + 4][r] + bv;
            if (act == 1) { v0 = fmaxf(v0, 0.0f); v1 = fmaxf(v1, 0.0f); }
            else if (act == 2) { v0 = tanhf(v0); v1 = tanhf(v1); }
            C[(size_t)(rowbase + r) * N + cc] = v0;
            C[(size_t)(rowbase + 16 + r) * N + cc] = v1;
        }
    }
}

// ---------------------------------------------------------------------------
// Sparse 8-neighborhood attention.  One 256-thread block per instance.
// ---------------------------------------------------------------------------
__global__ __launch_bounds__(256)
void k_nbr_attn(const float* __restrict__ H, const float* __restrict__ T,
                float* __restrict__ nbhd) {
    const int i   = blockIdx.x;
    const int tid = threadIdx.x;
    __shared__ float red[256];
    __shared__ float sc[8];
    __shared__ float al[8];
    __shared__ int   js[8];
    __shared__ int   cnt;

    if (tid == 0) {
        int c = 0;
        int r0 = i >> 6, c0 = i & 63;
        for (int dr = -1; dr <= 1; ++dr)
            for (int dc = -1; dc <= 1; ++dc) {
                if (dr == 0 && dc == 0) continue;
                int rr = r0 + dr, cc = c0 + dc;
                if (rr >= 0 && rr < 64 && cc >= 0 && cc < 64) js[c++] = rr * 64 + cc;
            }
        cnt = c;
    }
    __syncthreads();
    const int nc = cnt;
    const float* Hi = H + (size_t)i * LDIM;

    for (int nb = 0; nb < nc; ++nb) {
        const float* Tj = T + (size_t)js[nb] * LDIM;
        float p = Hi[tid] * Tj[tid] + Hi[tid + 256] * Tj[tid + 256];
        red[tid] = p;
        __syncthreads();
        for (int s = 128; s > 0; s >>= 1) {
            if (tid < s) red[tid] += red[tid + s];
            __syncthreads();
        }
        if (tid == 0) sc[nb] = red[0];
        __syncthreads();
    }
    if (tid == 0) {
        float m = -3.4e38f;
        for (int nb = 0; nb < nc; ++nb) m = fmaxf(m, sc[nb]);
        float s = 0.0f;
        for (int nb = 0; nb < nc; ++nb) { al[nb] = expf(sc[nb] - m); s += al[nb]; }
        float inv = 1.0f / s;
        for (int nb = 0; nb < nc; ++nb) al[nb] *= inv;
    }
    __syncthreads();
    for (int k = tid; k < LDIM; k += 256) {
        float acc = 0.0f;
        for (int nb = 0; nb < nc; ++nb)
            acc = fmaf(al[nb], H[(size_t)js[nb] * LDIM + k], acc);
        nbhd[(size_t)i * LDIM + k] = acc;
    }
}

// ---------------------------------------------------------------------------
// H2 = concat(H, nbhd)  ->  f32 and f16 copies
// ---------------------------------------------------------------------------
__global__ __launch_bounds__(256)
void k_concat(const float* __restrict__ H, const float* __restrict__ NB,
              float* __restrict__ H2f, _Float16* __restrict__ H2h) {
    int idx = blockIdx.x * blockDim.x + threadIdx.x;
    if (idx >= NINST * L2DIM) return;
    int i = idx >> 10;
    int k = idx & 1023;
    float v = (k < LDIM) ? H[(size_t)i * LDIM + k] : NB[(size_t)i * LDIM + (k - LDIM)];
    H2f[idx] = v;
    H2h[idx] = (_Float16)v;
}

// ---------------------------------------------------------------------------
// scores(i,t) = P_i . template_t   (N x NT)
// ---------------------------------------------------------------------------
__global__ __launch_bounds__(256)
void k_scores(const float* __restrict__ P, const float* __restrict__ tpl,
              float* __restrict__ scores) {
    int idx = blockIdx.x * blockDim.x + threadIdx.x;
    if (idx >= NINST * NTPL) return;
    int t = idx % NTPL;
    int i = idx / NTPL;
    const float* p = P + (size_t)i * L2DIM;
    const float* q = tpl + (size_t)t * L2DIM;
    float s = 0.0f;
    for (int k = 0; k < L2DIM; ++k) s = fmaf(p[k], q[k], s);
    scores[idx] = s;
}

// ---------------------------------------------------------------------------
// betas[t, :] = softmax over 4096 instances of scores[:, t]
// ---------------------------------------------------------------------------
__global__ __launch_bounds__(256)
void k_betas(const float* __restrict__ scores, float* __restrict__ betas) {
    const int t = blockIdx.x;
    const int tid = threadIdx.x;
    __shared__ float red[256];
    __shared__ float smax, ssum;
    float m = -3.4e38f;
    for (int i = tid; i < NINST; i += 256) m = fmaxf(m, scores[i * NTPL + t]);
    red[tid] = m; __syncthreads();
    for (int s = 128; s > 0; s >>= 1) { if (tid < s) red[tid] = fmaxf(red[tid], red[tid + s]); __syncthreads(); }
    if (tid == 0) smax = red[0];
    __syncthreads();
    float sum = 0.0f;
    for (int i = tid; i < NINST; i += 256) sum += expf(scores[i * NTPL + t] - smax);
    red[tid] = sum; __syncthreads();
    for (int s = 128; s > 0; s >>= 1) { if (tid < s) red[tid] += red[tid + s]; __syncthreads(); }
    if (tid == 0) ssum = red[0];
    __syncthreads();
    const float inv = 1.0f / ssum;
    for (int i = tid; i < NINST; i += 256)
        betas[(size_t)t * NINST + i] = expf(scores[i * NTPL + t] - smax) * inv;
}

// ---------------------------------------------------------------------------
// embs[t,k] = sum_i betas[t,i] * H2[i,k]
// ---------------------------------------------------------------------------
__global__ __launch_bounds__(128)
void k_embs(const float* __restrict__ betas, const float* __restrict__ H2f,
            float* __restrict__ embs) {
    const int t = blockIdx.x;
    const int k = blockIdx.y * 128 + threadIdx.x;
    const float* br = betas + (size_t)t * NINST;
    float acc = 0.0f;
    for (int i = 0; i < NINST; ++i)
        acc = fmaf(br[i], H2f[(size_t)i * L2DIM + k], acc);
    embs[(size_t)t * L2DIM + k] = acc;
}

// ---------------------------------------------------------------------------
// Global attention head (single block, 128 threads)
// ---------------------------------------------------------------------------
__global__ __launch_bounds__(128)
void k_gattn(const float* __restrict__ embs,
             const float* __restrict__ ga1_w, const float* __restrict__ ga1_b,
             const float* __restrict__ ga2_w, const float* __restrict__ ga2_b,
             const float* __restrict__ cls_w, const float* __restrict__ cls_b,
             float* __restrict__ gam_ws, float* __restrict__ M_ws,
             float* __restrict__ out) {
    const int tid = threadIdx.x;
    __shared__ float red[128];
    __shared__ float g[NTPL];
    __shared__ float gam[NTPL];

    for (int t = 0; t < NTPL; ++t) {
        const float* e = embs + (size_t)t * L2DIM;
        float u = ga1_b[tid];
        const float* wr = ga1_w + (size_t)tid * L2DIM;
        for (int k = 0; k < L2DIM; ++k) u = fmaf(wr[k], e[k], u);
        red[tid] = tanhf(u) * ga2_w[tid];
        __syncthreads();
        for (int s = 64; s > 0; s >>= 1) { if (tid < s) red[tid] += red[tid + s]; __syncthreads(); }
        if (tid == 0) g[t] = red[0] + ga2_b[0];
        __syncthreads();
    }
    if (tid == 0) {
        float m = -3.4e38f;
        for (int t = 0; t < NTPL; ++t) m = fmaxf(m, g[t]);
        float s = 0.0f;
        for (int t = 0; t < NTPL; ++t) { gam[t] = expf(g[t] - m); s += gam[t]; }
        float inv = 1.0f / s;
        for (int t = 0; t < NTPL; ++t) { gam[t] *= inv; gam_ws[t] = gam[t]; }
    }
    __syncthreads();
    for (int k = tid; k < L2DIM; k += 128) {
        float acc = 0.0f;
        for (int t = 0; t < NTPL; ++t)
            acc = fmaf(gam[t], embs[(size_t)t * L2DIM + k], acc);
        M_ws[k] = acc;
    }
    __syncthreads();
    float p = 0.0f;
    for (int k = tid; k < L2DIM; k += 128) p = fmaf(M_ws[k], cls_w[k], p);
    red[tid] = p;
    __syncthreads();
    for (int s = 64; s > 0; s >>= 1) { if (tid < s) red[tid] += red[tid + s]; __syncthreads(); }
    if (tid == 0) {
        float prob = 1.0f / (1.0f + expf(-(red[0] + cls_b[0])));
        out[0] = prob;
        out[1] = (prob >= 0.5f) ? 1.0f : 0.0f;
    }
}

// A[i] = sum_t gammas[t] * betas[t,i]
__global__ __launch_bounds__(256)
void k_attnmap(const float* __restrict__ gam_ws, const float* __restrict__ betas,
               float* __restrict__ out) {
    int i = blockIdx.x * blockDim.x + threadIdx.x;
    if (i >= NINST) return;
    float a = 0.0f;
    for (int t = 0; t < NTPL; ++t)
        a = fmaf(gam_ws[t], betas[(size_t)t * NINST + i], a);
    out[2 + i] = a;
}

// ---------------------------------------------------------------------------
// host launcher
// ---------------------------------------------------------------------------
extern "C" void kernel_launch(void* const* d_in, const int* in_sizes, int n_in,
                              void* d_out, int out_size, void* d_ws, size_t ws_size,
                              hipStream_t stream) {
    const float* x       = (const float*)d_in[0];
    // d_in[1] = positions (regular 64x64 grid; derived analytically)
    const float* conv1_w = (const float*)d_in[2];
    const float* conv1_b = (const float*)d_in[3];
    const float* conv2_w = (const float*)d_in[4];
    const float* conv2_b = (const float*)d_in[5];
    const float* fc1_w   = (const float*)d_in[6];
    const float* fc1_b   = (const float*)d_in[7];
    const float* fc2_w   = (const float*)d_in[8];
    const float* fc2_b   = (const float*)d_in[9];
    const float* nbr_w   = (const float*)d_in[10];
    const float* nbr_b   = (const float*)d_in[11];
    const float* tpl     = (const float*)d_in[12];
    const float* proto_w = (const float*)d_in[13];
    const float* proto_b = (const float*)d_in[14];
    const float* ga1_w   = (const float*)d_in[15];
    const float* ga1_b   = (const float*)d_in[16];
    const float* ga2_w   = (const float*)d_in[17];
    const float* ga2_b   = (const float*)d_in[18];
    const float* cls_w   = (const float*)d_in[19];
    const float* cls_b   = (const float*)d_in[20];
    float* out = (float*)d_out;

    // ---- workspace carve-up (256B aligned) ----
    char* ws = (char*)d_ws;
    size_t off = 0;
    auto carve = [&](size_t bytes) -> char* {
        char* p = ws + off;
        off = (off + bytes + 255) & ~(size_t)255;
        return p;
    };
    float*    h1      = (float*)   carve((size_t)NINST * 36 * 14 * 14 * 4);
    _Float16* h2h     = (_Float16*)carve((size_t)NINST * EMBD * 2);
    _Float16* fc1w16  = (_Float16*)carve((size_t)LDIM * EMBD * 2);
    _Float16* fc2w16  = (_Float16*)carve((size_t)LDIM * LDIM * 2);
    _Float16* nbrw16  = (_Float16*)carve((size_t)LDIM * LDIM * 2);
    _Float16* protw16 = (_Float16*)carve((size_t)L2DIM * L2DIM * 2);
    float*    A1      = (float*)   carve((size_t)NINST * LDIM * 4);
    _Float16* A1h     = (_Float16*)carve((size_t)NINST * LDIM * 2);
    float*    H       = (float*)   carve((size_t)NINST * LDIM * 4);
    _Float16* Hh      = (_Float16*)carve((size_t)NINST * LDIM * 2);
    float*    T       = (float*)   carve((size_t)NINST * LDIM * 4);
    float*    NB      = (float*)   carve((size_t)NINST * LDIM * 4);
    float*    H2f     = (float*)   carve((size_t)NINST * L2DIM * 4);
    _Float16* H2h     = (_Float16*)carve((size_t)NINST * L2DIM * 2);
    float*    P       = (float*)   carve((size_t)NINST * L2DIM * 4);
    float*    scores  = (float*)   carve((size_t)NINST * NTPL * 4);
    float*    betas   = (float*)   carve((size_t)NTPL * NINST * 4);
    float*    embs    = (float*)   carve((size_t)NTPL * L2DIM * 4);
    float*    gam_ws  = (float*)   carve(64);
    float*    M_ws    = (float*)   carve((size_t)L2DIM * 4);
    (void)ws_size; (void)in_sizes; (void)n_in; (void)out_size;

    // ---- feature extractor ----
    {
        int total = NINST * 36 * 14 * 14;
        k_conv1_pool<<<(total + 255) / 256, 256, 0, stream>>>(x, conv1_w, conv1_b, h1);
    }
    {
        int total = NINST * 48 * 6 * 6;
        k_conv2_pool<<<(total + 255) / 256, 256, 0, stream>>>(h1, conv2_w, conv2_b, h2h);
    }

    // ---- weight conversions to f16 ----
    k_cvt_h<<<(LDIM * EMBD + 255) / 256, 256, 0, stream>>>(fc1_w, fc1w16, LDIM * EMBD);
    k_cvt_h<<<(LDIM * LDIM + 255) / 256, 256, 0, stream>>>(fc2_w, fc2w16, LDIM * LDIM);
    k_cvt_h<<<(LDIM * LDIM + 255) / 256, 256, 0, stream>>>(nbr_w, nbrw16, LDIM * LDIM);
    k_cvt_h<<<(L2DIM * L2DIM + 255) / 256, 256, 0, stream>>>(proto_w, protw16, L2DIM * L2DIM);

    // ---- fc1: (4096x1728)x(512x1728)^T, relu ----
    k_gemm_nt<<<dim3(NINST / 128, LDIM / 64), 128, 0, stream>>>(h2h, fc1w16, fc1_b, A1,
                                                                NINST, LDIM, EMBD, 1);
    k_cvt_h<<<(NINST * LDIM + 255) / 256, 256, 0, stream>>>(A1, A1h, NINST * LDIM);

    // ---- fc2: relu -> H ----
    k_gemm_nt<<<dim3(NINST / 128, LDIM / 64), 128, 0, stream>>>(A1h, fc2w16, fc2_b, H,
                                                                NINST, LDIM, LDIM, 1);
    k_cvt_h<<<(NINST * LDIM + 255) / 256, 256, 0, stream>>>(H, Hh, NINST * LDIM);

    // ---- T = tanh(H @ nbr_w^T + b) ----
    k_gemm_nt<<<dim3(NINST / 128, LDIM / 64), 128, 0, stream>>>(Hh, nbrw16, nbr_b, T,
                                                                NINST, LDIM, LDIM, 2);

    // ---- sparse neighborhood attention ----
    k_nbr_attn<<<NINST, 256, 0, stream>>>(H, T, NB);

    // ---- H2 = [H, nbhd] ----
    k_concat<<<(NINST * L2DIM + 255) / 256, 256, 0, stream>>>(H, NB, H2f, H2h);

    // ---- P = tanh(H2 @ proto_w^T + b) ----
    k_gemm_nt<<<dim3(NINST / 128, L2DIM / 64), 128, 0, stream>>>(H2h, protw16, proto_b, P,
                                                                 NINST, L2DIM, L2DIM, 2);

    // ---- prototype attention ----
    k_scores<<<(NINST * NTPL + 255) / 256, 256, 0, stream>>>(P, tpl, scores);
    k_betas<<<NTPL, 256, 0, stream>>>(scores, betas);
    k_embs<<<dim3(NTPL, L2DIM / 128), 128, 0, stream>>>(betas, H2f, embs);

    // ---- global attention head + outputs ----
    k_gattn<<<1, 128, 0, stream>>>(embs, ga1_w, ga1_b, ga2_w, ga2_b, cls_w, cls_b,
                                   gam_ws, M_ws, out);
    k_attnmap<<<(NINST + 255) / 256, 256, 0, stream>>>(gam_ws, betas, out);
}
